// VidSam_63144609185927
// MI455X (gfx1250) — compile-verified, matrix operand order checked by reading, not run on hardware
//
#include <hip/hip_runtime.h>
#include <hip/hip_bf16.h>
#include <math.h>

// Problem constants (B=1): CK=64, CV=512, NO=3, T=4, H=W=64
#define CK    64
#define PQ    4096     // query positions  (H*W)
#define NM    16384    // memory positions (T*H*W)
#define MROWS 1536     // NO*CV output channels
#define KA    128      // packed K for the similarity GEMM (2*CK)
#define KSTEPS (NM / 32)

typedef __attribute__((ext_vector_type(16))) __bf16 v16bf;
typedef __attribute__((ext_vector_type(8)))  float  v8f;
typedef __attribute__((ext_vector_type(4)))  unsigned v4u;
typedef __attribute__((ext_vector_type(8)))  unsigned v8u;

// ---------- small helpers ----------
static __device__ __forceinline__ unsigned short f2bf(float f) {
    __bf16 h = (__bf16)f;
    return __builtin_bit_cast(unsigned short, h);
}
static __device__ __forceinline__ float bf2f(unsigned short s) {
    return (float)__builtin_bit_cast(__bf16, s);
}
// order-preserving float<->uint keys so we can use integer atomicMax
static __device__ __forceinline__ unsigned f2ord(float f) {
    unsigned u = __float_as_uint(f);
    return (u & 0x80000000u) ? ~u : (u | 0x80000000u);
}
static __device__ __forceinline__ float ord2f(unsigned u) {
    u = (u & 0x80000000u) ? (u ^ 0x80000000u) : ~u;
    return __uint_as_float(u);
}
static __device__ __forceinline__ v8f wmma_bf16(v16bf a, v16bf b, v8f c) {
    return __builtin_amdgcn_wmma_f32_16x16x32_bf16(false, a, false, b, (short)0, c,
                                                   false, false);
}
// cooperative (synchronous) load of a 128x32 bf16 tile into LDS (256 threads)
static __device__ __forceinline__ void load_tile(const unsigned short* __restrict__ src,
                                                 size_t row0, size_t ldk, size_t k0,
                                                 unsigned short* dst, int tid) {
#pragma unroll
    for (int i = 0; i < 2; ++i) {
        int chunk = tid + i * 256;
        int r = chunk >> 2, c = chunk & 3;
        uint4 v = *(const uint4*)(src + (row0 + (size_t)r) * ldk + k0 + (size_t)(c * 8));
        *(uint4*)(dst + r * 32 + c * 8) = v;
    }
}
// asynchronous global->LDS staging of a 128x32 bf16 tile (ASYNCcnt: 2 ops/wave)
static __device__ __forceinline__ void async_tile(const unsigned short* __restrict__ src,
                                                  size_t row0, size_t ldk, size_t k0,
                                                  unsigned short* dst, int tid) {
#pragma unroll
    for (int i = 0; i < 2; ++i) {
        int chunk = tid + i * 256;
        int r = chunk >> 2, c = chunk & 3;
        unsigned lds = (unsigned)(size_t)(dst + r * 32 + c * 8);   // low 32b = LDS offset
        const void* g = (const void*)(src + (row0 + (size_t)r) * ldk + k0 + (size_t)(c * 8));
        asm volatile("global_load_async_to_lds_b128 %0, %1, off"
                     :: "v"(lds), "v"(g) : "memory");
    }
}
static __device__ __forceinline__ void wait_async_le2() {
    asm volatile("s_wait_asynccnt 0x2" ::: "memory");
}
static __device__ __forceinline__ void wait_async_0() {
    asm volatile("s_wait_asynccnt 0x0" ::: "memory");
}
static __device__ __forceinline__ void wait_tensor(int le1) {
#if __has_builtin(__builtin_amdgcn_s_wait_tensorcnt)
    if (le1) __builtin_amdgcn_s_wait_tensorcnt(1);
    else     __builtin_amdgcn_s_wait_tensorcnt(0);
#else
    if (le1) asm volatile("s_wait_tensorcnt 0x1" ::: "memory");
    else     asm volatile("s_wait_tensorcnt 0x0" ::: "memory");
#endif
}
// Tensor Data Mover: DMA one 128-row x 32-col bf16 tile (row stride NM elems) into LDS.
// D# group0: count=1 | lds_addr[63:32] | global_addr[120:64] | type=2 [127:126]
// D# group1: data_size=2B, tile_dim0=32, tile_dim1=128, tensor_dim0_stride=16384,
//            tensor_dim0/1 large (no OOB), no padding / iterate / multicast.
static __device__ __forceinline__ void tdm_load_tile(const unsigned short* gsrc,
                                                     unsigned short* lds_dst) {
    unsigned lds = (unsigned)(size_t)lds_dst;
    unsigned long long ga = (unsigned long long)(size_t)gsrc;
    v4u g0 = { 1u,                                   // count = 1 (valid descriptor)
               lds,                                  // lds_addr
               (unsigned)ga,                         // global_addr[31:0]
               ((unsigned)(ga >> 32) & 0x1FFFFFFu) | 0x80000000u };  // ga[56:32] | type=2
    // tensor_dim0 = tensor_dim1 = 1<<20 (no OOB), tile_dim0 = 32, tile_dim1 = 128
    v8u g1 = { 0x00010000u,     // [17:16] data_size = 1 (2 bytes)
               0u,              // barrier addr 0 | tensor_dim0[15:0]<<16 (=0)
               16u,             // tensor_dim0[31:16]=16 | tensor_dim1[15:0]<<16 (=0)
               0x00200010u,     // tensor_dim1[31:16]=16 | tile_dim0(32)<<16
               128u,            // tile_dim1 = 128, tile_dim2 = 0
               16384u,          // tensor_dim0_stride low 32
               0u,              // stride hi | tensor_dim1_stride low
               0u };
    asm volatile("tensor_load_to_lds %0, %1" :: "s"(g0), "s"(g1) : "memory");
}
// A/B fragment: 16 rows x 32 K, one lane gets 16 contiguous bf16 from its row
static __device__ __forceinline__ v16bf frag(const unsigned short* t, int tile16, int lane) {
    int r  = tile16 * 16 + (lane & 15);
    int kh = (lane >> 4) * 16;
    return *(const v16bf*)(t + r * 32 + kh);
}

// ---------- prep kernels ----------
__global__ __launch_bounds__(256) void prep_q_kernel(
    const float* __restrict__ qk, const float* __restrict__ qe,
    unsigned short* __restrict__ Bbuf, float* __restrict__ bsq,
    unsigned* __restrict__ colmax, float* __restrict__ colsum)
{
    const int p = blockIdx.x * 256 + threadIdx.x;
    float bq = 0.f;
#pragma unroll 8
    for (int c = 0; c < CK; ++c) {
        float k = qk[c * PQ + p];
        float e = qe[c * PQ + p];
        Bbuf[(size_t)p * KA + 2 * c    ] = f2bf(e);
        Bbuf[(size_t)p * KA + 2 * c + 1] = f2bf(2.0f * k * e);
        bq += e * k * k;
    }
    bsq[p]    = bq;
    colmax[p] = 0x007FFFFFu;   // f2ord(-inf)
    colsum[p] = 0.f;
}

__global__ __launch_bounds__(256) void prep_mk_kernel(
    const float* __restrict__ mk, unsigned short* __restrict__ Abuf)
{
    const int n = blockIdx.x * 256 + threadIdx.x;
#pragma unroll 8
    for (int c = 0; c < CK; ++c) {
        float m = mk[(size_t)c * NM + n];
        Abuf[(size_t)n * KA + 2 * c    ] = f2bf(-m * m);
        Abuf[(size_t)n * KA + 2 * c + 1] = f2bf(m);
    }
}

__global__ __launch_bounds__(256) void prep_mv_kernel(
    const float4* __restrict__ mv, unsigned short* __restrict__ MVb)
{
    const size_t i = (size_t)blockIdx.x * 256 + threadIdx.x;
    float4 v = mv[i];
    ushort4 o;
    o.x = f2bf(v.x); o.y = f2bf(v.y); o.z = f2bf(v.z); o.w = f2bf(v.w);
    *(ushort4*)(MVb + 4 * i) = o;
}

// ---------- similarity GEMM (K=128): phase0 = column max, phase1 = write exp ----------
__global__ __launch_bounds__(256) void sim_kernel(
    const unsigned short* __restrict__ Abuf,   // [NM][KA] bf16
    const unsigned short* __restrict__ Bbuf,   // [PQ][KA] bf16
    const float* __restrict__ bsq,
    const float* __restrict__ ms,
    unsigned* __restrict__ colmax,
    float* __restrict__ colsum,
    unsigned short* __restrict__ Ebuf,         // [PQ][NM] bf16 (E^T)
    int phase)
{
    __shared__ __align__(128) unsigned short aT[128 * 32];
    __shared__ __align__(128) unsigned short bT[128 * 32];
    __shared__ __align__(128) unsigned short eT[128 * 128];

    const int tid = threadIdx.x, lane = tid & 31, w = tid >> 5;
    const int wm = w >> 1, wp = w & 1;
    const int rowBase = blockIdx.y * 128;       // n
    const int colBase = blockIdx.x * 128;       // p

    v8f acc[2][4] = {};
#pragma unroll
    for (int ks = 0; ks < 4; ++ks) {
        const int k0 = ks * 32;
        load_tile(Abuf, rowBase, KA, k0, aT, tid);
        load_tile(Bbuf, colBase, KA, k0, bT, tid);
        __syncthreads();
        v16bf a0 = frag(aT, wm * 2 + 0, lane);
        v16bf a1 = frag(aT, wm * 2 + 1, lane);
#pragma unroll
        for (int nt = 0; nt < 4; ++nt) {
            v16bf b = frag(bT, wp * 4 + nt, lane);
            acc[0][nt] = wmma_bf16(a0, b, acc[0][nt]);
            acc[1][nt] = wmma_bf16(a1, b, acc[1][nt]);
        }
        __syncthreads();
    }

    const int lhi = lane >> 4, lcol = lane & 15;
    if (phase == 0) {
#pragma unroll
        for (int nt = 0; nt < 4; ++nt) {
            const int p = colBase + wp * 64 + nt * 16 + lcol;
            const float bq = bsq[p];
            float cmax = -INFINITY;
#pragma unroll
            for (int mt = 0; mt < 2; ++mt) {
                const int mbase = rowBase + (wm * 2 + mt) * 16 + lhi * 8;
#pragma unroll
                for (int v = 0; v < 8; ++v) {
                    float s = (acc[mt][nt][v] - bq) * (ms[mbase + v] * 0.125f);
                    cmax = fmaxf(cmax, s);
                }
            }
            cmax = fmaxf(cmax, __shfl_xor(cmax, 16, 32));
            if (lhi == 0) atomicMax(&colmax[p], f2ord(cmax));
        }
    } else {
#pragma unroll
        for (int nt = 0; nt < 4; ++nt) {
            const int plocal = wp * 64 + nt * 16 + lcol;
            const int p  = colBase + plocal;
            const float bq = bsq[p];
            const float cm = ord2f(colmax[p]);
#pragma unroll
            for (int mt = 0; mt < 2; ++mt) {
                const int mloc  = (wm * 2 + mt) * 16 + lhi * 8;
                const int mbase = rowBase + mloc;
                unsigned wds[4];
#pragma unroll
                for (int v2 = 0; v2 < 4; ++v2) {
                    float s0 = (acc[mt][nt][2 * v2 + 0] - bq) * (ms[mbase + 2 * v2 + 0] * 0.125f);
                    float s1 = (acc[mt][nt][2 * v2 + 1] - bq) * (ms[mbase + 2 * v2 + 1] * 0.125f);
                    float e0 = __builtin_exp2f((s0 - cm) * 1.44269504f);
                    float e1 = __builtin_exp2f((s1 - cm) * 1.44269504f);
                    wds[v2] = (unsigned)f2bf(e0) | ((unsigned)f2bf(e1) << 16);
                }
                uint4 u; u.x = wds[0]; u.y = wds[1]; u.z = wds[2]; u.w = wds[3];
                *(uint4*)&eT[plocal * 128 + mloc] = u;
            }
        }
        __syncthreads();
        const int pl = tid >> 1, half = tid & 1;
        const int p = colBase + pl;
        const unsigned short* row = &eT[pl * 128 + half * 64];
        uint4* gdst = (uint4*)(Ebuf + (size_t)p * NM + rowBase + half * 64);
        float sum = 0.f;
#pragma unroll
        for (int i = 0; i < 8; ++i) gdst[i] = ((const uint4*)row)[i];
#pragma unroll
        for (int j = 0; j < 64; ++j) sum += bf2f(row[j]);
        atomicAdd(&colsum[p], sum);
    }
}

// ---------- readout GEMM: out[1536,4096] = MV x E / colsum ----------
// A (MV) tiles streamed by the Tensor Data Mover; B (E) tiles by async global->LDS.
// Double-buffered LDS, counter-based pipelining (TENSORcnt / ASYNCcnt).
__global__ __launch_bounds__(256) void readout_kernel(
    const unsigned short* __restrict__ MV,   // [MROWS][NM] bf16
    const unsigned short* __restrict__ E,    // [PQ][NM] bf16 (E^T)
    const float* __restrict__ colsum,
    float* __restrict__ out)                 // [MROWS][PQ]
{
    __shared__ __align__(128) unsigned short aT[2][128 * 32];
    __shared__ __align__(128) unsigned short bT[2][128 * 32];

    const int tid = threadIdx.x, lane = tid & 31, w = tid >> 5;
    const int wm = w >> 1, wp = w & 1;
    const int rowBase = blockIdx.y * 128;   // m (channel)
    const int colBase = blockIdx.x * 128;   // p

    // prologue: stage K-tile 0 into buffer 0
    if (w == 0) tdm_load_tile(MV + (size_t)rowBase * NM, aT[0]);
    async_tile(E, colBase, NM, 0, bT[0], tid);

    v8f acc[2][4] = {};
    for (int ks = 0; ks < KSTEPS; ++ks) {
        const int cur = ks & 1, nxt = cur ^ 1;
        if (ks + 1 < KSTEPS) {
            const size_t kn = (size_t)(ks + 1) * 32;
            if (w == 0) tdm_load_tile(MV + (size_t)rowBase * NM + kn, aT[nxt]);
            async_tile(E, colBase, NM, kn, bT[nxt], tid);
            wait_async_le2();            // retire current B tile, keep next in flight
            if (w == 0) wait_tensor(1);  // retire current A tile, keep next in flight
        } else {
            wait_async_0();
            if (w == 0) wait_tensor(0);
        }
        __syncthreads();                 // all waves' tiles visible

        v16bf a0 = frag(aT[cur], wm * 2 + 0, lane);
        v16bf a1 = frag(aT[cur], wm * 2 + 1, lane);
#pragma unroll
        for (int nt = 0; nt < 4; ++nt) {
            v16bf b = frag(bT[cur], wp * 4 + nt, lane);
            acc[0][nt] = wmma_bf16(a0, b, acc[0][nt]);
            acc[1][nt] = wmma_bf16(a1, b, acc[1][nt]);
        }
        __syncthreads();                 // done reading cur before it is refilled
    }

    const int lhi = lane >> 4, lcol = lane & 15;
#pragma unroll
    for (int nt = 0; nt < 4; ++nt) {
        const int p = colBase + wp * 64 + nt * 16 + lcol;
        const float inv = 1.0f / colsum[p];
#pragma unroll
        for (int mt = 0; mt < 2; ++mt) {
            const int m0 = rowBase + (wm * 2 + mt) * 16 + lhi * 8;
#pragma unroll
            for (int v = 0; v < 8; ++v)
                out[(size_t)(m0 + v) * PQ + p] = acc[mt][nt][v] * inv;
        }
    }
}

// ---------- launch ----------
extern "C" void kernel_launch(void* const* d_in, const int* in_sizes, int n_in,
                              void* d_out, int out_size, void* d_ws, size_t ws_size,
                              hipStream_t stream)
{
    (void)in_sizes; (void)n_in; (void)out_size; (void)ws_size;
    const float* qk = (const float*)d_in[0];
    const float* qe = (const float*)d_in[1];
    const float* mk = (const float*)d_in[2];
    const float* ms = (const float*)d_in[3];
    const float* mv = (const float*)d_in[4];
    float* out = (float*)d_out;
    char* ws = (char*)d_ws;

    unsigned short* Abuf = (unsigned short*)(ws);                 //  4 MiB
    unsigned short* Bbuf = (unsigned short*)(ws + (4u   << 20));  //  1 MiB
    unsigned short* MVb  = (unsigned short*)(ws + (5u   << 20));  // 48 MiB
    unsigned short* Ebuf = (unsigned short*)(ws + (53u  << 20));  // 128 MiB
    float*    bsq    = (float*)   (ws + (181u << 20));
    unsigned* colmax = (unsigned*)(ws + (181u << 20) + 16384);
    float*    colsum = (float*)   (ws + (181u << 20) + 32768);

    prep_q_kernel <<<PQ / 256, 256, 0, stream>>>(qk, qe, Bbuf, bsq, colmax, colsum);
    prep_mk_kernel<<<NM / 256, 256, 0, stream>>>(mk, Abuf);
    prep_mv_kernel<<<(MROWS * (NM / 4)) / 256, 256, 0, stream>>>((const float4*)mv, MVb);

    dim3 simGrid(PQ / 128, NM / 128);
    sim_kernel<<<simGrid, 256, 0, stream>>>(Abuf, Bbuf, bsq, ms, colmax, colsum, Ebuf, 0);
    sim_kernel<<<simGrid, 256, 0, stream>>>(Abuf, Bbuf, bsq, ms, colmax, colsum, Ebuf, 1);

    dim3 roGrid(PQ / 128, MROWS / 128);
    readout_kernel<<<roGrid, 256, 0, stream>>>(MVb, Ebuf, colsum, out);
}
